// ImprovedEmbeddingExpander_67267777790107
// MI455X (gfx1250) — compile-verified, matrix operand order checked by reading, not run on hardware
//
#include <hip/hip_runtime.h>

// ---------------------------------------------------------------------------
// CDNA5 (gfx1250): graph construction + 3-layer gated GCN.
// Heavy GEMMs: V_WMMA_F32_16X16X32_BF16, with the B panel cooperatively
// staged into LDS via GLOBAL_LOAD_ASYNC_TO_LDS_B128 (ASYNCcnt double-buffer).
// ---------------------------------------------------------------------------

typedef __attribute__((ext_vector_type(16))) __bf16        v16bf;
typedef __attribute__((ext_vector_type(8)))  float         v8f;
typedef __attribute__((ext_vector_type(4)))  unsigned int  v4u;

#define N_NODES 6000
#define N_PAD   6016      // 47 * 128 : padded M and K for the big GEMMs
#define NSUB    4500
#define BATCH   8
#define DIM     192
#define BD      1536      // BATCH * DIM
#define GD      40
#define KPQ     96        // 2*GD padded to multiple of 32
#define TOPK    10

__device__ __forceinline__ unsigned short f2bf(float f) {
  unsigned int u = __float_as_uint(f);
  u += 0x7FFFu + ((u >> 16) & 1u);          // round to nearest even
  return (unsigned short)(u >> 16);
}
__device__ __forceinline__ float bf2f(unsigned short s) {
  return __uint_as_float(((unsigned int)s) << 16);
}
__device__ __forceinline__ float softplus_f(float x) {
  if (x > 20.f) return x;
  return log1pf(expf(x));
}
__device__ __forceinline__ float sigmoid_f(float x) {
  return 1.f / (1.f + expf(-x));
}

// bf16 WMMA fragment: 8 contiguous bf16 at p (K-lo) + 8 at p+16 elems (K-hi).
__device__ __forceinline__ v16bf load_frag(const unsigned short* p) {
  union { v4u u[2]; v16bf v; } f;
  f.u[0] = *(const v4u*)p;
  f.u[1] = *(const v4u*)(p + 16);
  return f.v;
}

// Async copy 16 bytes global -> LDS (per lane), tracked by ASYNCcnt.
__device__ __forceinline__ void async_cp_b128(unsigned int ldsOff,
                                              const void* gsrc) {
  asm volatile("global_load_async_to_lds_b128 %0, %1, off"
               :: "v"(ldsOff), "v"((unsigned long long)gsrc) : "memory");
}
__device__ __forceinline__ void wait_async_le1() {
  asm volatile("s_wait_asynccnt 0x1" ::: "memory");
}
__device__ __forceinline__ void wait_async_le0() {
  asm volatile("s_wait_asynccnt 0x0" ::: "memory");
}

// ---------------------------------------------------------------------------
// Block = 8 waves = 128x64 macro-tile of C = [A0|A1](MxK row-major) * B
// (K x Ncols, column-major: element (k,n) at B[n*ldb + k]).
// Grid: (M/128, Ncols/64) -- exact, EXEC always full (WMMA requirement).
// B panel (64 cols x 32 k) is async-staged to LDS, double-buffered.
// EPI 0: fp32 out = softplus(0.5*acc) - 0.5                 (adjacency)
// EPI 1: bf16 out = acc                                      (SpMM aggregate)
// EPI 2: bf16 out = acc + bias[col]                          (mlp hn)
// EPI 3: g = sigmoid(acc+bias[col]); out = g*X + (1-g)*Y     (gate/diffusion)
// ---------------------------------------------------------------------------
template<int EPI>
__global__ __launch_bounds__(256)
void gemm_bf16(const unsigned short* __restrict__ A0,
               const unsigned short* __restrict__ A1,
               int lda, int kSplit,
               const unsigned short* __restrict__ Bm, int ldb, int K,
               void* __restrict__ OutP, int ldo,
               const float* __restrict__ bias,
               const unsigned short* __restrict__ X,
               const unsigned short* __restrict__ Y)
{
  __shared__ __align__(16) unsigned short bpanel[2][64 * 32]; // [col][k], 4KB each
  const int tid  = threadIdx.x;
  const int lane = tid & 31;
  const int wave = tid >> 5;
  const int m0   = (blockIdx.x * 8 + wave) * 16;
  const int n0   = blockIdx.y * 64;
  const int l15  = lane & 15;
  const int kOff = (lane >> 4) * 8;

  // cooperative B staging: thread t copies 16B; col = t>>2, chunk = (t&3)*8
  const int cpCol   = tid >> 2;
  const int cpChunk = (tid & 3) * 8;
  const unsigned short* bSrc = Bm + (size_t)(n0 + cpCol) * ldb + cpChunk;
  const unsigned int ldsDst0 =
      (unsigned int)(size_t)&bpanel[0][cpCol * 32 + cpChunk];
  const unsigned int ldsDst1 =
      (unsigned int)(size_t)&bpanel[1][cpCol * 32 + cpChunk];

  async_cp_b128(ldsDst0, bSrc);                 // prologue: panel for k0 = 0

  const unsigned short* aRow0 = A0 + (size_t)(m0 + l15) * lda;
  const unsigned short* aRow1 = A1 ? (A1 + (size_t)(m0 + l15) * lda) : aRow0;

  v8f acc[4];
#pragma unroll
  for (int t = 0; t < 4; ++t) acc[t] = (v8f){0.f,0.f,0.f,0.f,0.f,0.f,0.f,0.f};

  int par = 0;
  for (int k0 = 0; k0 < K; k0 += 32) {
    const int kn = k0 + 32;
    if (kn < K) {                               // issue next panel, then wait
      async_cp_b128(par ? ldsDst0 : ldsDst1, bSrc + kn);
      wait_async_le1();                         // current panel landed
    } else {
      wait_async_le0();
    }
    __syncthreads();                            // all waves' copies visible

    const unsigned short* ap = (k0 < kSplit) ? (aRow0 + k0 + kOff)
                                             : (aRow1 + (k0 - kSplit) + kOff);
    __builtin_prefetch(ap + 32, 0, 0);          // global_prefetch_b8
    v16bf a = load_frag(ap);
    const unsigned short* bp = &bpanel[par][0];
#pragma unroll
    for (int t = 0; t < 4; ++t) {               // ds_load_b128 x2 per fragment
      v16bf b = load_frag(bp + (t * 16 + l15) * 32 + kOff);
      acc[t] = __builtin_amdgcn_wmma_f32_16x16x32_bf16(
                 false, a, false, b, (short)0, acc[t], false, false);
    }
    __syncthreads();                            // panel reads done before reuse
    par ^= 1;
  }

  const int rBase = (lane >> 4) * 8;            // C layout: VGPR r -> M=r / r+8
#pragma unroll
  for (int t = 0; t < 4; ++t) {
    const int gn = n0 + t * 16 + l15;
#pragma unroll
    for (int r = 0; r < 8; ++r) {
      const int gm = m0 + rBase + r;
      const float v = acc[t][r];
      const size_t oi = (size_t)gm * ldo + gn;
      if constexpr (EPI == 0) {
        ((float*)OutP)[oi] = softplus_f(0.5f * v) - 0.5f;
      } else if constexpr (EPI == 1) {
        ((unsigned short*)OutP)[oi] = f2bf(v);
      } else if constexpr (EPI == 2) {
        ((unsigned short*)OutP)[oi] = f2bf(v + bias[gn]);
      } else {
        const float g  = sigmoid_f(v + bias[gn]);
        const float xv = bf2f(X[oi]);
        const float yv = bf2f(Y[oi]);
        ((unsigned short*)OutP)[oi] = f2bf(g * xv + (1.f - g) * yv);
      }
    }
  }
}

// ------------------------- small helper kernels ----------------------------

__global__ void fill_u16(unsigned short* p, long n) {
  long i = (long)blockIdx.x * blockDim.x + threadIdx.x;
  if (i < n) p[i] = 0;
}
__global__ void fill_i32(int* p, int n, int val) {
  int i = blockIdx.x * blockDim.x + threadIdx.x;
  if (i < n) p[i] = val;
}

// nv1 = tanh(3*(emb1@W1+b1)), nv2 likewise; pack P=[nv1|nv2], Q=[nv2|nv1]
__global__ void node_vec_kernel(const float* __restrict__ emb1,
                                const float* __restrict__ emb2,
                                const float* __restrict__ l1w, const float* __restrict__ l1b,
                                const float* __restrict__ l2w, const float* __restrict__ l2b,
                                unsigned short* __restrict__ P,
                                unsigned short* __restrict__ Q)
{
  int idx = blockIdx.x * blockDim.x + threadIdx.x;
  if (idx >= N_NODES * GD) return;
  int i = idx / GD, k = idx - i * GD;
  float s1 = l1b[k], s2 = l2b[k];
#pragma unroll 8
  for (int t = 0; t < GD; ++t) {
    s1 += emb1[i * GD + t] * l1w[t * GD + k];
    s2 += emb2[i * GD + t] * l2w[t * GD + k];
  }
  float v1 = tanhf(3.0f * s1);
  float v2 = tanhf(3.0f * s2);
  P[(size_t)i * KPQ + k]      = f2bf(v1);
  P[(size_t)i * KPQ + GD + k] = f2bf(v2);
  Q[(size_t)i * KPQ + k]      = f2bf(v2);
  Q[(size_t)i * KPQ + GD + k] = f2bf(v1);
}

// fp32 row-major (KxD) weight -> bf16 column-major (dst[c*K+k])
__global__ void wconv_kernel(const float* __restrict__ src,
                             unsigned short* __restrict__ dst, int K, int D)
{
  int idx = blockIdx.x * blockDim.x + threadIdx.x;
  if (idx >= K * D) return;
  int k = idx / D, c = idx - k * D;
  dst[(size_t)c * K + k] = f2bf(src[(size_t)k * D + c]);
}

// Per row: 10 threshold-descent max passes -> top-10 cutoff, then masked
// relu row-sum (+1 for identity) -> dis = rsqrt(degree).
__global__ __launch_bounds__(256)
void topk_kernel(const float* __restrict__ adj, int pitch,
                 float* __restrict__ thr, float* __restrict__ dis)
{
  __shared__ float red[256];
  const int row = blockIdx.x;
  const int tid = threadIdx.x;
  const float* rp = adj + (size_t)row * pitch;
  float cur = 3.0e38f;
  for (int t = 0; t < TOPK; ++t) {
    float loc = -3.0e38f;
    for (int j = tid; j < N_NODES; j += 256) {
      float v = rp[j];
      if (v < cur && v > loc) loc = v;
    }
    red[tid] = loc; __syncthreads();
    for (int s = 128; s > 0; s >>= 1) {
      if (tid < s) red[tid] = fmaxf(red[tid], red[tid + s]);
      __syncthreads();
    }
    cur = red[0]; __syncthreads();
  }
  float acc = 0.f;
  for (int j = tid; j < N_NODES; j += 256) {
    float v = rp[j];
    if (v >= cur) acc += fmaxf(v, 0.f);
  }
  red[tid] = acc; __syncthreads();
  for (int s = 128; s > 0; s >>= 1) {
    if (tid < s) red[tid] += red[tid + s];
    __syncthreads();
  }
  if (tid == 0) {
    thr[row] = cur;
    dis[row] = rsqrtf(red[0] + 1.0f);
  }
}

// adjT[m][n] = normAdj[n][m] = (mask_relu(adj[n][m]) + (n==m)) * dis[n]*dis[m]
// bf16, full N_PAD x N_PAD coverage (pad rows/cols zeroed). LDS-tiled.
__global__ __launch_bounds__(256)
void norm_transpose_kernel(const float* __restrict__ adj,
                           const float* __restrict__ thr,
                           const float* __restrict__ dis,
                           unsigned short* __restrict__ adjT)
{
  __shared__ unsigned short tile[32][33];
  const int mBase = blockIdx.x * 32;   // adjT row  (adj col)
  const int nBase = blockIdx.y * 32;   // adjT col  (adj row)
  const int tx = threadIdx.x, ty = threadIdx.y;
#pragma unroll
  for (int k = 0; k < 32; k += 8) {
    const int nn = nBase + ty + k;
    const int mm = mBase + tx;
    float v = 0.f;
    if (nn < N_NODES && mm < N_NODES) {
      float raw = adj[(size_t)nn * N_PAD + mm];
      float mv  = (raw >= thr[nn]) ? fmaxf(raw, 0.f) : 0.f;
      if (nn == mm) mv += 1.f;
      v = mv * dis[nn] * dis[mm];
    }
    tile[tx][ty + k] = f2bf(v);
  }
  __syncthreads();
#pragma unroll
  for (int k = 0; k < 32; k += 8) {
    adjT[(size_t)(mBase + ty + k) * N_PAD + (nBase + tx)] = tile[ty + k][tx];
  }
}

// h (6000 x 1536, bf16 row-major) -> hT (1536 x 6016 col-major B operand)
__global__ __launch_bounds__(256)
void transpose_h_kernel(const unsigned short* __restrict__ h,
                        unsigned short* __restrict__ hT)
{
  __shared__ unsigned short tile[32][33];
  const int jBase = blockIdx.x * 32;   // column of h (< 1536)
  const int nBase = blockIdx.y * 32;   // row of h (< 6016, padded)
  const int tx = threadIdx.x, ty = threadIdx.y;
#pragma unroll
  for (int k = 0; k < 32; k += 8) {
    const int nn = nBase + ty + k;
    tile[tx][ty + k] = (nn < N_NODES) ? h[(size_t)nn * BD + (jBase + tx)]
                                      : (unsigned short)0;
  }
  __syncthreads();
#pragma unroll
  for (int k = 0; k < 32; k += 8) {
    hT[(size_t)(jBase + ty + k) * N_PAD + (nBase + tx)] = tile[ty + k][tx];
  }
}

__global__ void scatter_pos_kernel(const int* __restrict__ idxsub,
                                   int* __restrict__ pos, int ns)
{
  int s = blockIdx.x * blockDim.x + threadIdx.x;
  if (s < ns) pos[idxsub[s]] = s;
}

// h0[n][b][c] = observed ? subset[b][pos[n]][c] : unobserved_init[c]  (bf16)
__global__ void init_h0_kernel(const float* __restrict__ subset,
                               const float* __restrict__ uinit,
                               const int* __restrict__ pos,
                               unsigned short* __restrict__ h)
{
  long idx = (long)blockIdx.x * blockDim.x + threadIdx.x;
  if (idx >= (long)N_NODES * BD) return;
  int n = (int)(idx / BD);
  int j = (int)(idx - (long)n * BD);
  int b = j / DIM, c = j - b * DIM;
  int p = pos[n];
  float v = (p >= 0) ? subset[((size_t)b * NSUB + p) * DIM + c] : uinit[c];
  h[idx] = f2bf(v);
}

// out[b][n][c] = observed ? (1-a)*subset + a*h  :  h     (fp32 output)
__global__ void final_kernel(const unsigned short* __restrict__ h,
                             const float* __restrict__ subset,
                             const int* __restrict__ pos,
                             const float* __restrict__ alpha_p,
                             float* __restrict__ out)
{
  long idx = (long)blockIdx.x * blockDim.x + threadIdx.x;
  if (idx >= (long)BATCH * N_NODES * DIM) return;
  int b = (int)(idx / ((long)N_NODES * DIM));
  long rem = idx - (long)b * N_NODES * DIM;
  int n = (int)(rem / DIM);
  int c = (int)(rem - (long)n * DIM);
  float hv = bf2f(h[((size_t)n * BATCH + b) * DIM + c]);
  int p = pos[n];
  float o = hv;
  if (p >= 0) {
    float a  = alpha_p[0];
    float h0 = subset[((size_t)b * NSUB + p) * DIM + c];
    o = (1.f - a) * h0 + a * hv;
  }
  out[idx] = o;
}

// ---------------------------------------------------------------------------

static inline int cdiv_i(long a, long b) { return (int)((a + b - 1) / b); }

extern "C" void kernel_launch(void* const* d_in, const int* in_sizes, int n_in,
                              void* d_out, int out_size, void* d_ws, size_t ws_size,
                              hipStream_t stream)
{
  (void)in_sizes; (void)n_in; (void)out_size; (void)ws_size;
  const float* subset = (const float*)d_in[0];
  const int*   idxsub = (const int*)  d_in[1];
  const float* emb1   = (const float*)d_in[2];
  const float* emb2   = (const float*)d_in[3];
  const float* l1w    = (const float*)d_in[4];
  const float* l1b    = (const float*)d_in[5];
  const float* l2w    = (const float*)d_in[6];
  const float* l2b    = (const float*)d_in[7];
  const float* uinit  = (const float*)d_in[8];
  const float* mlp_w  = (const float*)d_in[9];
  const float* mlp_b  = (const float*)d_in[10];
  const float* gate_w = (const float*)d_in[11];
  const float* gate_b = (const float*)d_in[12];
  const float* diff_w = (const float*)d_in[13];
  const float* diff_b = (const float*)d_in[14];
  const float* alpha  = (const float*)d_in[15];
  float* out = (float*)d_out;

  // ---- workspace carve-up (~350 MB) ----
  char* w = (char*)d_ws;
  size_t off = 0;
  auto carve = [&](size_t bytes) -> void* {
    void* p = w + off;
    off = (off + bytes + 255) & ~(size_t)255;
    return p;
  };
  unsigned short* Pbf   = (unsigned short*)carve((size_t)N_PAD * KPQ * 2);
  unsigned short* Qbf   = (unsigned short*)carve((size_t)N_PAD * KPQ * 2);
  float*          adj   = (float*)carve((size_t)N_PAD * N_PAD * 4);
  float*          thr   = (float*)carve((size_t)N_NODES * 4);
  float*          dis   = (float*)carve((size_t)N_NODES * 4);
  int*            pos   = (int*)  carve((size_t)N_NODES * 4);
  unsigned short* adjT  = (unsigned short*)carve((size_t)N_PAD * N_PAD * 2);
  unsigned short* hA    = (unsigned short*)carve((size_t)N_NODES * BD * 2);
  unsigned short* hB    = (unsigned short*)carve((size_t)N_NODES * BD * 2);
  unsigned short* hn    = (unsigned short*)carve((size_t)N_NODES * BD * 2);
  unsigned short* lo    = (unsigned short*)carve((size_t)N_NODES * BD * 2);
  unsigned short* agg   = (unsigned short*)carve((size_t)N_PAD * BD * 2);
  unsigned short* hT    = (unsigned short*)carve((size_t)BD * N_PAD * 2);
  unsigned short* wmlp  = (unsigned short*)carve((size_t)3 * DIM * DIM * 2);
  unsigned short* wgate = (unsigned short*)carve((size_t)3 * 2 * DIM * DIM * 2);
  unsigned short* wdiff = (unsigned short*)carve((size_t)3 * 2 * DIM * DIM * 2);

  // ---- stage 1: node vectors (P/Q zero-padded in K and M) ----
  {
    long np = (long)N_PAD * KPQ;
    fill_u16<<<cdiv_i(np, 256), 256, 0, stream>>>(Pbf, np);
    fill_u16<<<cdiv_i(np, 256), 256, 0, stream>>>(Qbf, np);
  }
  node_vec_kernel<<<cdiv_i((long)N_NODES * GD, 256), 256, 0, stream>>>(
      emb1, emb2, l1w, l1b, l2w, l2b, Pbf, Qbf);

  // ---- weights -> bf16 column-major ----
  for (int i = 0; i < 3; ++i) {
    wconv_kernel<<<cdiv_i(DIM * DIM, 256), 256, 0, stream>>>(
        mlp_w + (size_t)i * DIM * DIM, wmlp + (size_t)i * DIM * DIM, DIM, DIM);
    wconv_kernel<<<cdiv_i(2 * DIM * DIM, 256), 256, 0, stream>>>(
        gate_w + (size_t)i * 2 * DIM * DIM, wgate + (size_t)i * 2 * DIM * DIM, 2 * DIM, DIM);
    wconv_kernel<<<cdiv_i(2 * DIM * DIM, 256), 256, 0, stream>>>(
        diff_w + (size_t)i * 2 * DIM * DIM, wdiff + (size_t)i * 2 * DIM * DIM, 2 * DIM, DIM);
  }

  // ---- stage 2: symmetric similarity GEMM (K=96) + softplus epilogue ----
  gemm_bf16<0><<<dim3(N_PAD / 128, N_PAD / 64), 256, 0, stream>>>(
      Pbf, nullptr, KPQ, KPQ, Qbf, KPQ, KPQ,
      adj, N_PAD, nullptr, nullptr, nullptr);

  // ---- stage 3: top-k threshold + degree, then normalize+transpose ----
  topk_kernel<<<N_NODES, 256, 0, stream>>>(adj, N_PAD, thr, dis);
  norm_transpose_kernel<<<dim3(N_PAD / 32, N_PAD / 32), dim3(32, 8), 0, stream>>>(
      adj, thr, dis, adjT);

  // ---- stage 4: h0 scatter (layout [n][b][c], bf16) ----
  fill_i32<<<cdiv_i(N_NODES, 256), 256, 0, stream>>>(pos, N_NODES, -1);
  scatter_pos_kernel<<<cdiv_i(NSUB, 256), 256, 0, stream>>>(idxsub, pos, NSUB);
  init_h0_kernel<<<cdiv_i((long)N_NODES * BD, 256), 256, 0, stream>>>(
      subset, uinit, pos, hA);

  // ---- stage 5: 3 gated graph-conv layers ----
  unsigned short* hc = hA;
  unsigned short* hx = hB;
  for (int i = 0; i < 3; ++i) {
    transpose_h_kernel<<<dim3(BD / 32, N_PAD / 32), dim3(32, 8), 0, stream>>>(hc, hT);
    // agg = normAdj^T @ H      (M=K=6016, L2-resident adjacency)
    gemm_bf16<1><<<dim3(N_PAD / 128, BD / 64), 256, 0, stream>>>(
        adjT, nullptr, N_PAD, N_PAD, hT, N_PAD, N_PAD,
        agg, BD, nullptr, nullptr, nullptr);
    // hn = agg @ mlp_w + mlp_b        (M = 48000 = 375*128)
    gemm_bf16<2><<<dim3(375, DIM / 64), 256, 0, stream>>>(
        agg, nullptr, DIM, DIM, wmlp + (size_t)i * DIM * DIM, DIM, DIM,
        hn, DIM, mlp_b + (size_t)i * DIM, nullptr, nullptr);
    // layer_out = sig([h|hn]@gate_w+b)*hn + (1-g)*h
    gemm_bf16<3><<<dim3(375, DIM / 64), 256, 0, stream>>>(
        hc, hn, DIM, DIM, wgate + (size_t)i * 2 * DIM * DIM, 2 * DIM, 2 * DIM,
        lo, DIM, gate_b + (size_t)i * DIM, hn, hc);
    // h_new = sig([h|layer_out]@diff_w+b)*layer_out + (1-dg)*h
    gemm_bf16<3><<<dim3(375, DIM / 64), 256, 0, stream>>>(
        hc, lo, DIM, DIM, wdiff + (size_t)i * 2 * DIM * DIM, 2 * DIM, 2 * DIM,
        hx, DIM, diff_b + (size_t)i * DIM, lo, hc);
    unsigned short* t = hc; hc = hx; hx = t;
  }

  // ---- stage 6: confidence fusion, fp32 output in [b][n][c] ----
  final_kernel<<<cdiv_i((long)BATCH * N_NODES * DIM, 256), 256, 0, stream>>>(
      hc, subset, pos, alpha, out);
}
